// TopoLinear_47656957116758
// MI455X (gfx1250) — matrix operand order repaired and checked by reading, not compiled, not run on hardware
//
#include <hip/hip_runtime.h>
#include <math.h>

typedef __attribute__((ext_vector_type(16))) __bf16 v16bf;
typedef __attribute__((ext_vector_type(8)))  float  v8f;

#define Bq 8
#define Nn 1024
#define Cc 384
#define Dd 384
#define Ss 128

// ---------------------------------------------------------------------------
// wn = scale * l2norm(maps_weight) rows              (128 x 128)
// ---------------------------------------------------------------------------
__global__ void wn_kernel(const float* __restrict__ mw,
                          const float* __restrict__ scale,
                          float* __restrict__ wn) {
  __shared__ float red[128];
  int s = blockIdx.x, t = threadIdx.x;
  float v = mw[s * 128 + t];
  red[t] = v * v;
  __syncthreads();
  for (int o = 64; o > 0; o >>= 1) {
    if (t < o) red[t] += red[t + o];
    __syncthreads();
  }
  float inv = scale[0] / fmaxf(sqrtf(red[0]), 1e-12f);
  wn[s * 128 + t] = v * inv;
}

// ---------------------------------------------------------------------------
// coef[n] = softmax( l2norm(pos_embed[n]) @ wn^T )   (1024 x 128)
// ---------------------------------------------------------------------------
__global__ void coef_kernel(const float* __restrict__ pos,
                            const float* __restrict__ wn,
                            float* __restrict__ coef) {
  __shared__ float pe[128];
  __shared__ float red[128];
  int n = blockIdx.x, t = threadIdx.x;
  float v = pos[n * 128 + t];
  pe[t] = v;
  red[t] = v * v;
  __syncthreads();
  for (int o = 64; o > 0; o >>= 1) {
    if (t < o) red[t] += red[t + o];
    __syncthreads();
  }
  float inv = 1.0f / fmaxf(sqrtf(red[0]), 1e-12f);
  __syncthreads();
  const float* wr = wn + t * 128;
  float dot = 0.f;
  #pragma unroll 8
  for (int p = 0; p < 128; ++p) dot += pe[p] * wr[p];
  dot *= inv;
  red[t] = dot;
  __syncthreads();
  for (int o = 64; o > 0; o >>= 1) {
    if (t < o) red[t] = fmaxf(red[t], red[t + o]);
    __syncthreads();
  }
  float m = red[0];
  __syncthreads();
  float e = __expf(dot - m);
  red[t] = e;
  __syncthreads();
  for (int o = 64; o > 0; o >>= 1) {
    if (t < o) red[t] += red[t + o];
    __syncthreads();
  }
  coef[n * 128 + t] = e / red[0];
}

// ---------------------------------------------------------------------------
// bout[n][d] = sum_s coef[n][s] * bias[d][s]         (1024 x 384)
// ---------------------------------------------------------------------------
__global__ void bout_kernel(const float* __restrict__ coef,
                            const float* __restrict__ bias,
                            float* __restrict__ bout) {
  __shared__ float cs[128];
  int n = blockIdx.x, t = threadIdx.x;
  cs[t] = coef[n * 128 + t];
  __syncthreads();
  #pragma unroll
  for (int i = 0; i < 3; ++i) {
    int d = t + i * 128;
    const float* br = bias + d * 128;
    float a = 0.f;
    #pragma unroll 8
    for (int s = 0; s < 128; ++s) a += cs[s] * br[s];
    bout[n * 384 + d] = a;
  }
}

// ---------------------------------------------------------------------------
// weight f32 -> bf16, identity layout (s contiguous => b128 operand loads)
// ---------------------------------------------------------------------------
__global__ void cvtw_kernel(const float* __restrict__ src,
                            __bf16* __restrict__ dst, int count) {
  int i = blockIdx.x * blockDim.x + threadIdx.x;
  int stride = gridDim.x * blockDim.x;
  for (; i < count; i += stride) dst[i] = (__bf16)src[i];
}

// ---------------------------------------------------------------------------
// f32 -> bf16 with WMMA A-operand layout per 32-element chunk:
//   chunk positions [0..31]=e{0..31} (contiguous k order)
// so each lane-half reads its whole 16-element A fragment as one v16bf.
// Used for both x (rows of length 384) and coef (rows of length 128).
// ---------------------------------------------------------------------------
__global__ void pk_kernel(const float* __restrict__ src,
                          __bf16* __restrict__ dst, int count, int L) {
  int i = blockIdx.x * blockDim.x + threadIdx.x;
  int stride = gridDim.x * blockDim.x;
  for (; i < count; i += stride) {
    int row = i / L;
    int c   = i - row * L;
    int q   = c >> 5;         // 32-wide chunk
    int r   = c & 31;
    int g   = r >> 3;         // group of 8: 0,1,2,3 -> 0,1,2,3
    int gp  = g;
    int pos = (gp << 3) | (r & 7);
    dst[(size_t)row * L + (q << 5) + pos] = (__bf16)src[i];
  }
}

// ---------------------------------------------------------------------------
// Fused main kernel: per wave = 16 tokens x 16 outputs (d).
//   combine (WMMA, K=S=128):  Wt[n,d,c] = sum_s coef[n,s]*weight[d,c,s]
//   stage 32-c chunk of Wt in LDS (bf16), then
//   apply   (WMMA, K=32 c's): out[b,n,d] += sum_c x[b,n,c]*Wt[n,d,c]
// ---------------------------------------------------------------------------
__global__ __launch_bounds__(256) void topo_main(
    const __bf16* __restrict__ cpk,    // (N, S) bf16, A-operand permuted
    const __bf16* __restrict__ wbf,    // (D, C, S) bf16
    const __bf16* __restrict__ xpk,    // (B, N, C) bf16, A-operand permuted
    const float*  __restrict__ bout,   // (N, D) f32
    float* __restrict__ out) {         // (B, N, D) f32
  __shared__ __bf16 stage[8][16 * 16 * 32];  // 16 KB per wave: Wt [n][d][c32]

  const int wave = threadIdx.x >> 5;
  const int lane = threadIdx.x & 31;
  const int dl = lane & 15;
  const int hi = lane >> 4;
  const int t0 = blockIdx.x * 16;               // token tile base
  const int d0 = (blockIdx.y * 8 + wave) * 16;  // d tile base
  __bf16* st = stage[wave];

  // --- resident A operand: coef tile (16 tokens x 128 s), one load per k ---
  v16bf A[4];
  #pragma unroll
  for (int k = 0; k < 4; ++k) {
    A[k] = *(const v16bf*)(cpk + (size_t)(t0 + dl) * Ss + k * 32 + hi * 16);
  }

  // --- persistent output accumulators: one 16x16 tile (b x d) per token ----
  v8f acc[16];
  #pragma unroll
  for (int n = 0; n < 16; ++n) acc[n] = {};

  for (int c0 = 0; c0 < Cc; c0 += 32) {
    // ---------------- combine: build Wt[n][d][c0..c0+31] in LDS ------------
    #pragma unroll 4
    for (int ci = 0; ci < 32; ++ci) {
      const int c = c0 + ci;
      // B operand: rows K = s, cols N = d; lanes 0-15 K lo-16, 16-31 K hi-16.
      const __bf16* bp = wbf + ((size_t)(d0 + dl) * Cc + c) * Ss + hi * 16;
      v8f w = {};
      #pragma unroll
      for (int k = 0; k < 4; ++k) {
        v16bf Bv = *(const v16bf*)(bp + k * 32);
        w = __builtin_amdgcn_wmma_f32_16x16x32_bf16(
            false, A[k], false, Bv, (short)0, w, false, false);
      }
      // C layout: lane dl = col d; rows n = j + 8*hi.  Scatter to LDS.
      #pragma unroll
      for (int j = 0; j < 8; ++j) {
        const int n = j + 8 * hi;
        st[(n * 16 + dl) * 32 + ci] = (__bf16)w[j];
      }
    }

    // ---------------- apply: one full-K WMMA per token ---------------------
    #pragma unroll
    for (int n = 0; n < 16; ++n) {
      // A operand rows = batch b.  Rows 8-15 of D are never stored, so lanes
      // dl>=8 may carry duplicate (finite) data instead of zeros: load
      // unconditionally with b = dl&7 — no branch, no zero-fill.
      const __bf16* xp =
          xpk + ((size_t)(dl & 7) * Nn + t0 + n) * Cc + c0 + hi * 16;
      v16bf Ax = *(const v16bf*)xp;
      // B operand: rows K = c-local (32), cols = d.  One 32B LDS load.
      v16bf Bx = *(const v16bf*)&st[(n * 16 + dl) * 32 + hi * 16];
      acc[n] = __builtin_amdgcn_wmma_f32_16x16x32_bf16(
          false, Ax, false, Bx, (short)0, acc[n], false, false);
    }
  }

  // --- epilogue: rows b=0..7 live in lanes 0-15 (vgpr j = b) ---------------
  if (hi == 0) {
    #pragma unroll
    for (int n = 0; n < 16; ++n) {
      const float bv = bout[(size_t)(t0 + n) * Dd + d0 + dl];
      #pragma unroll
      for (int j = 0; j < 8; ++j) {
        out[((size_t)j * Nn + t0 + n) * Dd + d0 + dl] = acc[n][j] + bv;
      }
    }
  }
}

// ---------------------------------------------------------------------------
extern "C" void kernel_launch(void* const* d_in, const int* in_sizes, int n_in,
                              void* d_out, int out_size, void* d_ws,
                              size_t ws_size, hipStream_t stream) {
  (void)in_sizes; (void)n_in; (void)out_size; (void)ws_size;
  const float* x      = (const float*)d_in[0];  // (8,1024,384)
  const float* pos    = (const float*)d_in[1];  // (1024,128)
  const float* mw     = (const float*)d_in[2];  // (128,128)
  const float* scale  = (const float*)d_in[3];  // ()
  const float* weight = (const float*)d_in[4];  // (384,384,128)
  const float* bias   = (const float*)d_in[5];  // (384,128)
  float* out = (float*)d_out;

  char* ws = (char*)d_ws;
  float*  coef = (float*)(ws + 0);           //   524288 B
  float*  wn   = (float*)(ws + 524288);      //    65536 B
  float*  bout = (float*)(ws + 589824);      //  1572864 B
  __bf16* wbf  = (__bf16*)(ws + 2162688);    // 37748736 B
  __bf16* xpk  = (__bf16*)(ws + 39911424);   //  6291456 B
  __bf16* cpk  = (__bf16*)(ws + 46202880);   //   262144 B  (total ~44.3 MB)

  wn_kernel<<<128, 128, 0, stream>>>(mw, scale, wn);
  coef_kernel<<<1024, 128, 0, stream>>>(pos, wn, coef);
  bout_kernel<<<1024, 128, 0, stream>>>(coef, bias, bout);
  cvtw_kernel<<<4608, 256, 0, stream>>>(weight, wbf, Dd * Cc * Ss);
  pk_kernel<<<1536, 256, 0, stream>>>(x, xpk, Bq * Nn * Cc, Cc);
  pk_kernel<<<512, 256, 0, stream>>>(coef, cpk, Nn * Ss, Ss);
  topo_main<<<dim3(64, 3), 256, 0, stream>>>(cpk, wbf, xpk, bout, out);
}
